// ConformerBlock_12360915878711
// MI455X (gfx1250) — matrix operand order repaired and checked by reading, not compiled, not run on hardware
//
#include <hip/hip_runtime.h>
#include <hip/hip_bf16.h>
#include <math.h>

// ---------------- types / helpers ----------------
typedef __attribute__((ext_vector_type(16))) __bf16 v16bf;
typedef __attribute__((ext_vector_type(8)))  float  v8f;
typedef __attribute__((ext_vector_type(4)))  unsigned v4u;
typedef __attribute__((ext_vector_type(8)))  int      v8i;
typedef __attribute__((ext_vector_type(4)))  int      v4i;

#define DEV __device__ __forceinline__

DEV __bf16 f2bf(float f) {
  unsigned u = __builtin_bit_cast(unsigned, f);
  unsigned r = u + 0x7FFFu + ((u >> 16) & 1u);            // RNE
  unsigned short h = (unsigned short)(r >> 16);
  return __builtin_bit_cast(__bf16, h);
}
DEV float bf2f(__bf16 b) {
  unsigned short h = __builtin_bit_cast(unsigned short, b);
  unsigned u = ((unsigned)h) << 16;
  return __builtin_bit_cast(float, u);
}
DEV float sigmoidf_(float x) { return 1.0f / (1.0f + __expf(-x)); }
DEV float siluf_(float x)    { return x * sigmoidf_(x); }

DEV unsigned lds_off(const void* p) { return (unsigned)(size_t)p; }

// ---------------- Tensor Data Mover: 2D bf16 tile -> LDS ----------------
// D# per CDNA5 ISA §8: group0 {count=1, lds_addr, global_addr(57b), type=2},
// group1 {data_size=2B, pad_enable, pad_interval/amount, tensor_dim0/1,
//         tile_dim0/1, tensor_dim0_stride}. OOB rows/cols zero-fill.
DEV void tdm_load_2d(const void* gptr, unsigned lds_byte,
                     unsigned tensor_d0, unsigned tensor_d1,
                     unsigned tile_d0, unsigned tile_d1,
                     unsigned stride_d0_elems,
                     unsigned pad_interval_code, unsigned pad_amount_code)
{
  unsigned long long ga = (unsigned long long)(size_t)gptr;
  v4u g0;
  g0[0] = 1u;                                             // count=1 (valid), user mode
  g0[1] = lds_byte;                                       // LDS dest (bytes)
  g0[2] = (unsigned)(ga & 0xFFFFFFFFu);                   // global_addr[31:0]
  g0[3] = (unsigned)((ga >> 32) & 0x01FFFFFFu) | (2u << 30); // addr[56:32] | type=2
  v8i g1;
  unsigned w0 = (1u << 16)                                // data_size: 2 bytes
              | (1u << 20)                                // pad_enable (LDS row pad)
              | (pad_interval_code << 22)
              | (pad_amount_code << 25);
  g1[0] = (int)w0;
  g1[1] = (int)((tensor_d0 & 0xFFFFu) << 16);             // abar_addr=0 | dim0[15:0]
  g1[2] = (int)(((tensor_d0 >> 16) & 0xFFFFu) | ((tensor_d1 & 0xFFFFu) << 16));
  g1[3] = (int)(((tensor_d1 >> 16) & 0xFFFFu) | (tile_d0 << 16));
  g1[4] = (int)(tile_d1 & 0xFFFFu);                       // tile_dim1 | tile_dim2=0
  g1[5] = (int)stride_d0_elems;                           // dim0_stride[31:0]
  g1[6] = 0;                                              // dim0_stride[47:32] | d1s lo
  g1[7] = 0;
  v4i z4 = {0, 0, 0, 0};
#if __clang_major__ >= 23
  v8i z8 = {0, 0, 0, 0, 0, 0, 0, 0};
  __builtin_amdgcn_tensor_load_to_lds(g0, g1, z4, z4, z8, 0);
#else
  __builtin_amdgcn_tensor_load_to_lds(g0, g1, z4, z4, 0);
#endif
}

// ---------------- problem constants ----------------
constexpr int B_  = 4;
constexpr int T_  = 2048;
constexpr int H_  = 512;
constexpr int NH_ = 8;
constexpr int D_  = 64;
constexpr int I_  = 2048;
constexpr int KW_ = 31;
constexpr int L_  = 2 * T_ - 1;     // 4095
constexpr int M_  = B_ * T_;        // 8192

// ---------------- fp32 -> bf16 convert ----------------
__global__ void cvt_k(const float* __restrict__ src, __bf16* __restrict__ dst, int n) {
  int i = blockIdx.x * blockDim.x + threadIdx.x;
  if (i < n) dst[i] = f2bf(src[i]);
}

// ---------------- LayerNorm over H=512 (block per row, 128 thr) ----------------
template<bool OUT_BF16>
__global__ __launch_bounds__(128)
void layernorm_k(const float* __restrict__ x, const float* __restrict__ g,
                 const float* __restrict__ b, void* __restrict__ outp)
{
  const int row  = blockIdx.x;
  const int tid  = threadIdx.x;
  const int lane = tid & 31, wid = tid >> 5;
  const float* xr = x + (size_t)row * H_;
  float4 v = *(const float4*)&xr[tid * 4];
  float s  = v.x + v.y + v.z + v.w;
  float s2 = v.x * v.x + v.y * v.y + v.z * v.z + v.w * v.w;
  for (int o = 16; o >= 1; o >>= 1) { s += __shfl_xor(s, o); s2 += __shfl_xor(s2, o); }
  __shared__ float rs_[4], rq_[4];
  if (lane == 0) { rs_[wid] = s; rq_[wid] = s2; }
  __syncthreads();
  s  = rs_[0] + rs_[1] + rs_[2] + rs_[3];
  s2 = rq_[0] + rq_[1] + rq_[2] + rq_[3];
  const float mean = s * (1.0f / (float)H_);
  const float var  = s2 * (1.0f / (float)H_) - mean * mean;
  const float rstd = rsqrtf(var + 1e-5f);
  const int c = tid * 4;
  float4 gg = *(const float4*)&g[c];
  float4 bb = *(const float4*)&b[c];
  float y0 = (v.x - mean) * rstd * gg.x + bb.x;
  float y1 = (v.y - mean) * rstd * gg.y + bb.y;
  float y2 = (v.z - mean) * rstd * gg.z + bb.z;
  float y3 = (v.w - mean) * rstd * gg.w + bb.w;
  if (OUT_BF16) {
    __bf16* o2 = (__bf16*)outp + (size_t)row * H_ + c;
    o2[0] = f2bf(y0); o2[1] = f2bf(y1); o2[2] = f2bf(y2); o2[3] = f2bf(y3);
  } else {
    float* o2 = (float*)outp + (size_t)row * H_ + c;
    float4 yo; yo.x = y0; yo.y = y1; yo.z = y2; yo.w = y3;
    *(float4*)o2 = yo;
  }
}

// ---------------- WMMA bf16 GEMM: C[M,N] = epi( A[M,K] @ W[N,K]^T ) ----------------
// Block = 128 thr (4 waves). Block tile 64x64, wave tile 32x32 (2x2 WMMA 16x16).
// Tiles staged by the Tensor Data Mover, double-buffered on TENSORcnt.
// LDS rows are 48 elems (96B): TDM pads 8 DWORDs after every 16 DWORDs (64B row).
enum { EPI_F32 = 0, EPI_SILU_BF16 = 1, EPI_BF16 = 2, EPI_RES_F32 = 3 };

template<int EPI>
__global__ __launch_bounds__(128)
void gemm_bf16_k(const __bf16* __restrict__ A, const __bf16* __restrict__ W,
                 float* __restrict__ outF, __bf16* __restrict__ outB,
                 const float* __restrict__ res, float scale,
                 int M, int N, int K)
{
  __shared__ __bf16 As[2][64][48];
  __shared__ __bf16 Ws[2][64][48];
  const int tid  = threadIdx.x;
  const int lane = tid & 31, wid = tid >> 5;
  const int lm = lane & 15, kh = lane >> 4;
  const int mBase = blockIdx.y * 64, nBase = blockIdx.x * 64;
  const int wm = (wid >> 1) * 32, wn = (wid & 1) * 32;
  v8f acc[2][2] = {};
  union UV { uint4 q[2]; v16bf v; };
  const int nIter = K >> 5;

  if (wid == 0) {   // prologue: stage k0=0 into buffer 0
    tdm_load_2d(A + (size_t)mBase * K, lds_off(&As[0][0][0]),
                (unsigned)K, (unsigned)(M - mBase), 32, 64, (unsigned)K, 3, 7);
    tdm_load_2d(W + (size_t)nBase * K, lds_off(&Ws[0][0][0]),
                (unsigned)K, (unsigned)(N - nBase), 32, 64, (unsigned)K, 3, 7);
  }
  for (int it = 0; it < nIter; ++it) {
    const int cur = it & 1;
    if (wid == 0) __builtin_amdgcn_s_wait_tensorcnt(0);
    __syncthreads();
    asm volatile("" ::: "memory");
    if (wid == 0 && it + 1 < nIter) {                 // issue next while computing
      const int k0n = (it + 1) << 5;
      tdm_load_2d(A + (size_t)mBase * K + k0n, lds_off(&As[cur ^ 1][0][0]),
                  (unsigned)(K - k0n), (unsigned)(M - mBase), 32, 64, (unsigned)K, 3, 7);
      tdm_load_2d(W + (size_t)nBase * K + k0n, lds_off(&Ws[cur ^ 1][0][0]),
                  (unsigned)(K - k0n), (unsigned)(N - nBase), 32, 64, (unsigned)K, 3, 7);
    }
    // A fragments: lane half kh holds K = kh*16 .. kh*16+15 (contiguous)
    v16bf af[2], bfg[2];
    for (int i = 0; i < 2; ++i) {
      UV u; int rr = wm + i * 16 + lm;
      u.q[0] = *(const uint4*)&As[cur][rr][kh * 16];
      u.q[1] = *(const uint4*)&As[cur][rr][kh * 16 + 8];
      af[i] = u.v;
    }
    // B fragments: lane half kh holds K = kh*16 .. kh*16+15 (contiguous)
    for (int j = 0; j < 2; ++j) {
      UV u; int rr = wn + j * 16 + lm;
      u.q[0] = *(const uint4*)&Ws[cur][rr][kh * 16];
      u.q[1] = *(const uint4*)&Ws[cur][rr][kh * 16 + 8];
      bfg[j] = u.v;
    }
    for (int i = 0; i < 2; ++i)
      for (int j = 0; j < 2; ++j)
        acc[i][j] = __builtin_amdgcn_wmma_f32_16x16x32_bf16(
            false, af[i], false, bfg[j], (short)0, acc[i][j], false, false);
  }
  // C layout: lane holds (row = r + 8*(lane>>4), col = lane&15) in VGPR r
  for (int i = 0; i < 2; ++i)
    for (int j = 0; j < 2; ++j)
      for (int r = 0; r < 8; ++r) {
        int gm = mBase + wm + i * 16 + r + 8 * kh;
        if (gm >= M) continue;
        int gn = nBase + wn + j * 16 + lm;
        float v = acc[i][j][r];
        size_t idx = (size_t)gm * N + gn;
        if (EPI == EPI_F32)            outF[idx] = v;
        else if (EPI == EPI_SILU_BF16) outB[idx] = f2bf(siluf_(v));
        else if (EPI == EPI_BF16)      outB[idx] = f2bf(v);
        else                           outF[idx] = res[idx] + scale * v;
      }
}

// ---------------- Flash attention with fused Transformer-XL rel-shift ----------------
// Grid: (T/64, NH, B). Block 128 thr = 4 waves, each wave owns 16 query rows.
// Key blocks of 32. Rel-shift identity: scores[t,s] uses rel index l = (T-1)+s-t.
// Q/K/V and the 96-row rel-k band are staged by TDM (72-elem LDS rows:
// pad 4 DWORDs after every 32 DWORDs; OOB band rows hardware-zero-filled).
__global__ __launch_bounds__(128)
void attn_k(const __bf16* __restrict__ Q, const __bf16* __restrict__ Kp,
            const __bf16* __restrict__ Vp, const __bf16* __restrict__ Rp,
            const float* __restrict__ bias_u, const float* __restrict__ bias_v,
            __bf16* __restrict__ Ctx)
{
  __shared__ __bf16 Qs[64][72];
  __shared__ __bf16 Ks[32][72];
  __shared__ __bf16 Vs[32][72];
  __shared__ __bf16 Rs[96][72];
  __shared__ float  Pp[4][16][52];    // per-wave scratch: pos panel, then probs
  const int tid  = threadIdx.x;
  const int lane = tid & 31, w = tid >> 5;
  const int lm = lane & 15, kh = lane >> 4;
  const int b = blockIdx.z, h = blockIdx.y, t0 = blockIdx.x * 64;
  const size_t qbase = ((size_t)(b * T_ + t0)) * H_ + h * D_;

  if (tid < 32) {
    tdm_load_2d(Q + qbase, lds_off(&Qs[0][0]), 64, 64, 64, 64, (unsigned)H_, 4, 3);
    __builtin_amdgcn_s_wait_tensorcnt(0);
  }
  __syncthreads();
  asm volatile("" ::: "memory");

  // q+bias_u / q+bias_v A-fragments (16 x 64, two K=32 frags each)
  v16bf qu[2], qv[2];
  for (int f = 0; f < 2; ++f)
    for (int e = 0; e < 16; ++e) {
      int d = f * 32 + kh * 16 + e;
      float qf = bf2f(Qs[16 * w + lm][d]);
      qu[f][e] = f2bf(qf + bias_u[h * D_ + d]);
      qv[f][e] = f2bf(qf + bias_v[h * D_ + d]);
    }

  float m_i[8], l_i[8];
  v8f o[4] = {};
  for (int r = 0; r < 8; ++r) { m_i[r] = -__builtin_inff(); l_i[r] = 0.f; }
  const int rofs = 48 - 16 * w;   // wave's offset into the 96-row rel band
  union UV { uint4 q[2]; v16bf v; };

  for (int s0 = 0; s0 < T_; s0 += 32) {
    __syncthreads();              // previous iteration's tile reads complete
    const size_t kbase = ((size_t)(b * T_ + s0)) * H_ + h * D_;
    const int lbase0 = (T_ - 1) + s0 - t0 - 63;   // >= 0 always
    if (tid < 32) {
      tdm_load_2d(Kp + kbase, lds_off(&Ks[0][0]), 64, (unsigned)(T_ - s0),
                  64, 32, (unsigned)H_, 4, 3);
      tdm_load_2d(Vp + kbase, lds_off(&Vs[0][0]), 64, (unsigned)(T_ - s0),
                  64, 32, (unsigned)H_, 4, 3);
      tdm_load_2d(Rp + (size_t)lbase0 * H_ + h * D_, lds_off(&Rs[0][0]),
                  64, (unsigned)(L_ - lbase0), 64, 96, (unsigned)H_, 4, 3);
      __builtin_amdgcn_s_wait_tensorcnt(0);
    }
    __syncthreads();
    asm volatile("" ::: "memory");

    // content scores: (q+bu) @ K^T -> [16 x 32]
    v8f sc[2] = {};
    for (int f = 0; f < 2; ++f)
      for (int j = 0; j < 2; ++j) {
        UV u; int rr = j * 16 + lm;
        u.q[0] = *(const uint4*)&Ks[rr][f * 32 + kh * 16];
        u.q[1] = *(const uint4*)&Ks[rr][f * 32 + kh * 16 + 8];
        sc[j] = __builtin_amdgcn_wmma_f32_16x16x32_bf16(
            false, qu[f], false, u.v, (short)0, sc[j], false, false);
      }
    // pos panel: (q+bv) @ relk^T over 48-wide band -> [16 x 48]
    v8f pp[3] = {};
    for (int f = 0; f < 2; ++f)
      for (int jt = 0; jt < 3; ++jt) {
        UV u; int rr = rofs + jt * 16 + lm;
        u.q[0] = *(const uint4*)&Rs[rr][f * 32 + kh * 16];
        u.q[1] = *(const uint4*)&Rs[rr][f * 32 + kh * 16 + 8];
        pp[jt] = __builtin_amdgcn_wmma_f32_16x16x32_bf16(
            false, qv[f], false, u.v, (short)0, pp[jt], false, false);
      }
    for (int jt = 0; jt < 3; ++jt)
      for (int r = 0; r < 8; ++r)
        Pp[w][r + 8 * kh][jt * 16 + lm] = pp[jt][r];
    asm volatile("s_wait_dscnt 0x0" ::: "memory");

    // gather rel-shift diagonal (col = s-t+15), scale, online softmax
    for (int r = 0; r < 8; ++r) {
      int m = r + 8 * kh;
      float v0 = (sc[0][r] + Pp[w][m][lm - m + 15]) * 0.125f;
      float v1 = (sc[1][r] + Pp[w][m][16 + lm - m + 15]) * 0.125f;
      float mx = fmaxf(v0, v1);
      for (int ofx = 8; ofx >= 1; ofx >>= 1) mx = fmaxf(mx, __shfl_xor(mx, ofx));
      float mnew = fmaxf(m_i[r], mx);
      float corr = __expf(m_i[r] - mnew);
      float p0 = __expf(v0 - mnew), p1 = __expf(v1 - mnew);
      float rs = p0 + p1;
      for (int ofx = 8; ofx >= 1; ofx >>= 1) rs += __shfl_xor(rs, ofx);
      l_i[r] = l_i[r] * corr + rs;
      m_i[r] = mnew;
      for (int jd = 0; jd < 4; ++jd) o[jd][r] *= corr;
      Pp[w][m][lm]      = p0;    // C-layout -> A-layout transpose via LDS
      Pp[w][m][16 + lm] = p1;
    }
    asm volatile("s_wait_dscnt 0x0" ::: "memory");

    v16bf pf;
    for (int e = 0; e < 16; ++e) {
      int kk = kh * 16 + e;
      pf[e] = f2bf(Pp[w][lm][kk]);
    }
    // P @ V -> accumulate [16 x 64]
    for (int jd = 0; jd < 4; ++jd) {
      v16bf bv_;
      for (int e = 0; e < 16; ++e) bv_[e] = Vs[kh * 16 + e][jd * 16 + lm];
      o[jd] = __builtin_amdgcn_wmma_f32_16x16x32_bf16(
          false, pf, false, bv_, (short)0, o[jd], false, false);
    }
  }

  for (int jd = 0; jd < 4; ++jd)
    for (int r = 0; r < 8; ++r) {
      int t = t0 + 16 * w + r + 8 * kh;
      int d = jd * 16 + lm;
      Ctx[((size_t)(b * T_ + t)) * H_ + h * D_ + d] = f2bf(o[jd][r] / l_i[r]);
    }
}

// ---------------- GLU over channel split ----------------
__global__ void glu_k(const float* __restrict__ y, float* __restrict__ out, int M)
{
  size_t idx = (size_t)blockIdx.x * blockDim.x + threadIdx.x;
  if (idx >= (size_t)M * H_) return;
  size_t m = idx >> 9; int c = (int)(idx & (H_ - 1));
  float a  = y[m * (2 * H_) + c];
  float gg = y[m * (2 * H_) + H_ + c];
  out[idx] = a * sigmoidf_(gg);
}

// ---------------- depthwise conv (K=31, pad 15) + BN + SiLU ----------------
__global__ void dwconv_bn_silu_k(const float* __restrict__ in, const float* __restrict__ dw,
                                 const float* __restrict__ bg, const float* __restrict__ bb,
                                 const float* __restrict__ bm, const float* __restrict__ bv,
                                 __bf16* __restrict__ out)
{
  size_t idx = (size_t)blockIdx.x * blockDim.x + threadIdx.x;
  if (idx >= (size_t)B_ * T_ * H_) return;
  int c = (int)(idx % H_);
  int t = (int)((idx / H_) % T_);
  int b = (int)(idx / ((size_t)H_ * T_));
  const float* base = in + (size_t)b * T_ * H_;
  float acc = 0.f;
#pragma unroll
  for (int k = 0; k < KW_; ++k) {
    int tt = t + k - (KW_ / 2);
    if (tt >= 0 && tt < T_) acc += base[(size_t)tt * H_ + c] * dw[c * KW_ + k];
  }
  float y = (acc - bm[c]) * rsqrtf(bv[c] + 1e-5f) * bg[c] + bb[c];
  out[idx] = f2bf(siluf_(y));
}

// ---------------- host orchestration ----------------
extern "C" void kernel_launch(void* const* d_in, const int* in_sizes, int n_in,
                              void* d_out, int out_size, void* d_ws, size_t ws_size,
                              hipStream_t stream)
{
  (void)in_sizes; (void)n_in; (void)out_size; (void)ws_size;
  const float* x     = (const float*)d_in[0];
  const float* pos   = (const float*)d_in[1];
  const float* ln1g  = (const float*)d_in[2];
  const float* ln1b  = (const float*)d_in[3];
  const float* ff1w1 = (const float*)d_in[4];
  const float* ff1w2 = (const float*)d_in[5];
  const float* ln2g  = (const float*)d_in[6];
  const float* ln2b  = (const float*)d_in[7];
  const float* wq    = (const float*)d_in[8];
  const float* wk    = (const float*)d_in[9];
  const float* wv    = (const float*)d_in[10];
  const float* wo    = (const float*)d_in[11];
  const float* wrel  = (const float*)d_in[12];
  const float* bu    = (const float*)d_in[13];
  const float* bvv   = (const float*)d_in[14];
  const float* ln3g  = (const float*)d_in[15];
  const float* ln3b  = (const float*)d_in[16];
  const float* pw1   = (const float*)d_in[17];
  const float* dwv   = (const float*)d_in[18];
  const float* bng   = (const float*)d_in[19];
  const float* bnb   = (const float*)d_in[20];
  const float* bnm   = (const float*)d_in[21];
  const float* bnv   = (const float*)d_in[22];
  const float* pw2   = (const float*)d_in[23];
  const float* ln4g  = (const float*)d_in[24];
  const float* ln4b  = (const float*)d_in[25];
  const float* ff2w1 = (const float*)d_in[26];
  const float* ff2w2 = (const float*)d_in[27];
  const float* ln5g  = (const float*)d_in[28];
  const float* ln5b  = (const float*)d_in[29];

  size_t off = 0;
  auto alloc = [&](size_t bytes) -> void* {
    off = (off + 255) & ~(size_t)255;
    void* p = (char*)d_ws + off;
    off += bytes;
    return p;
  };
  // bf16 weight mirrors
  __bf16* wb_ff1w1 = (__bf16*)alloc((size_t)I_ * H_ * 2);
  __bf16* wb_ff1w2 = (__bf16*)alloc((size_t)H_ * I_ * 2);
  __bf16* wb_q     = (__bf16*)alloc((size_t)H_ * H_ * 2);
  __bf16* wb_k     = (__bf16*)alloc((size_t)H_ * H_ * 2);
  __bf16* wb_v     = (__bf16*)alloc((size_t)H_ * H_ * 2);
  __bf16* wb_o     = (__bf16*)alloc((size_t)H_ * H_ * 2);
  __bf16* wb_rel   = (__bf16*)alloc((size_t)H_ * H_ * 2);
  __bf16* wb_pw1   = (__bf16*)alloc((size_t)2 * H_ * H_ * 2);
  __bf16* wb_pw2   = (__bf16*)alloc((size_t)H_ * H_ * 2);
  __bf16* wb_ff2w1 = (__bf16*)alloc((size_t)I_ * H_ * 2);
  __bf16* wb_ff2w2 = (__bf16*)alloc((size_t)H_ * I_ * 2);
  __bf16* pe_bf    = (__bf16*)alloc((size_t)4096 * H_ * 2);
  // activations
  __bf16* lnbuf   = (__bf16*)alloc((size_t)M_ * H_ * 2);
  __bf16* h1      = (__bf16*)alloc((size_t)M_ * I_ * 2);
  __bf16* qb      = (__bf16*)alloc((size_t)M_ * H_ * 2);
  __bf16* kb      = (__bf16*)alloc((size_t)M_ * H_ * 2);
  __bf16* vb      = (__bf16*)alloc((size_t)M_ * H_ * 2);
  __bf16* relkb   = (__bf16*)alloc((size_t)4096 * H_ * 2);
  __bf16* ctx     = (__bf16*)alloc((size_t)M_ * H_ * 2);
  float*  xcur    = (float*) alloc((size_t)M_ * H_ * 4);
  float*  pw1out  = (float*) alloc((size_t)M_ * 2 * H_ * 4);
  float*  gluout  = (float*) alloc((size_t)M_ * H_ * 4);
  __bf16* convout = (__bf16*)alloc((size_t)M_ * H_ * 2);

  auto cvt = [&](const float* s, __bf16* d, int n) {
    cvt_k<<<(n + 255) / 256, 256, 0, stream>>>(s, d, n);
  };
  cvt(ff1w1, wb_ff1w1, I_ * H_);
  cvt(ff1w2, wb_ff1w2, H_ * I_);
  cvt(wq, wb_q, H_ * H_);   cvt(wk, wb_k, H_ * H_);
  cvt(wv, wb_v, H_ * H_);   cvt(wo, wb_o, H_ * H_);
  cvt(wrel, wb_rel, H_ * H_);
  cvt(pw1, wb_pw1, 2 * H_ * H_);
  cvt(pw2, wb_pw2, H_ * H_);
  cvt(ff2w1, wb_ff2w1, I_ * H_);
  cvt(ff2w2, wb_ff2w2, H_ * I_);
  cvt(pos, pe_bf, L_ * H_);

  const dim3 blk(128);
  // ---- FFN1: x += 0.5 * silu(ln1(x) W1^T) W2^T ----
  layernorm_k<true><<<M_, blk, 0, stream>>>(x, ln1g, ln1b, lnbuf);
  gemm_bf16_k<EPI_SILU_BF16><<<dim3(I_ / 64, M_ / 64), blk, 0, stream>>>(
      lnbuf, wb_ff1w1, nullptr, h1, nullptr, 0.f, M_, I_, H_);
  gemm_bf16_k<EPI_RES_F32><<<dim3(H_ / 64, M_ / 64), blk, 0, stream>>>(
      h1, wb_ff1w2, xcur, nullptr, x, 0.5f, M_, H_, I_);
  // ---- attention ----
  layernorm_k<true><<<M_, blk, 0, stream>>>(xcur, ln2g, ln2b, lnbuf);
  gemm_bf16_k<EPI_BF16><<<dim3(H_ / 64, M_ / 64), blk, 0, stream>>>(
      lnbuf, wb_q, nullptr, qb, nullptr, 0.f, M_, H_, H_);
  gemm_bf16_k<EPI_BF16><<<dim3(H_ / 64, M_ / 64), blk, 0, stream>>>(
      lnbuf, wb_k, nullptr, kb, nullptr, 0.f, M_, H_, H_);
  gemm_bf16_k<EPI_BF16><<<dim3(H_ / 64, M_ / 64), blk, 0, stream>>>(
      lnbuf, wb_v, nullptr, vb, nullptr, 0.f, M_, H_, H_);
  gemm_bf16_k<EPI_BF16><<<dim3(H_ / 64, (L_ + 63) / 64), blk, 0, stream>>>(
      pe_bf, wb_rel, nullptr, relkb, nullptr, 0.f, L_, H_, H_);
  attn_k<<<dim3(T_ / 64, NH_, B_), blk, 0, stream>>>(qb, kb, vb, relkb, bu, bvv, ctx);
  gemm_bf16_k<EPI_RES_F32><<<dim3(H_ / 64, M_ / 64), blk, 0, stream>>>(
      ctx, wb_o, xcur, nullptr, xcur, 1.0f, M_, H_, H_);
  // ---- conv module ----
  layernorm_k<true><<<M_, blk, 0, stream>>>(xcur, ln3g, ln3b, lnbuf);
  gemm_bf16_k<EPI_F32><<<dim3((2 * H_) / 64, M_ / 64), blk, 0, stream>>>(
      lnbuf, wb_pw1, pw1out, nullptr, nullptr, 0.f, M_, 2 * H_, H_);
  glu_k<<<(int)(((size_t)M_ * H_ + 255) / 256), 256, 0, stream>>>(pw1out, gluout, M_);
  dwconv_bn_silu_k<<<(int)(((size_t)B_ * T_ * H_ + 255) / 256), 256, 0, stream>>>(
      gluout, dwv, bng, bnb, bnm, bnv, convout);
  gemm_bf16_k<EPI_RES_F32><<<dim3(H_ / 64, M_ / 64), blk, 0, stream>>>(
      convout, wb_pw2, xcur, nullptr, xcur, 1.0f, M_, H_, H_);
  // ---- FFN2 ----
  layernorm_k<true><<<M_, blk, 0, stream>>>(xcur, ln4g, ln4b, lnbuf);
  gemm_bf16_k<EPI_SILU_BF16><<<dim3(I_ / 64, M_ / 64), blk, 0, stream>>>(
      lnbuf, wb_ff2w1, nullptr, h1, nullptr, 0.f, M_, I_, H_);
  gemm_bf16_k<EPI_RES_F32><<<dim3(H_ / 64, M_ / 64), blk, 0, stream>>>(
      h1, wb_ff2w2, xcur, nullptr, xcur, 0.5f, M_, H_, I_);
  // ---- final LN -> d_out (fp32) ----
  layernorm_k<false><<<M_, blk, 0, stream>>>(xcur, ln5g, ln5b, d_out);
}
